// Attention_6734508720503
// MI455X (gfx1250) — matrix operand order repaired and checked by reading, not compiled, not run on hardware
//
#include <hip/hip_runtime.h>
#include <hip/hip_bf16.h>

// CDNA5 / gfx1250 wave32 WMMA implementation of channel attention
// (1x1 conv -> 3x3 dwconv -> per-head channel attention -> 1x1 conv).
// Matmuls: v_wmma_f32_16x16x32_bf16 (bf16 operands, f32 accum).
// Data movement: async global->LDS staging (ASYNCcnt) where no transpose or
// dtype conversion is needed; LDS vector-fragment reads elsewhere.

typedef __attribute__((ext_vector_type(16))) __bf16 v16bf;
typedef __attribute__((ext_vector_type(8)))  __bf16 v8bf;
typedef __attribute__((ext_vector_type(8)))  float  v8f;
typedef __attribute__((ext_vector_type(4)))  int    v4i;

static __device__ inline v8f zero8() { v8f z = {}; return z; }

static __device__ inline v16bf cat8(v8bf lo, v8bf hi8) {
  return __builtin_shufflevector(lo, hi8, 0, 1, 2, 3, 4, 5, 6, 7, 8, 9, 10, 11,
                                 12, 13, 14, 15);
}

// ---- async global -> LDS (gfx1250 ASYNCcnt path), with sync fallback ------
#if defined(__has_builtin)
#if __has_builtin(__builtin_amdgcn_global_load_async_to_lds_b128)
#define HAVE_ASYNC_LDS 1
#endif
#endif
#ifndef HAVE_ASYNC_LDS
#define HAVE_ASYNC_LDS 0
#endif

#if HAVE_ASYNC_LDS
static __device__ inline void async_copy16(const void* g, void* l) {
  // builtin signature (from toolchain diagnostic): pointers to 16B int vectors
  __builtin_amdgcn_global_load_async_to_lds_b128((v4i*)g, (v4i*)l, 0, 0);
}
static __device__ inline void async_wait0() {
#if __has_builtin(__builtin_amdgcn_s_wait_asynccnt)
  __builtin_amdgcn_s_wait_asynccnt(0);
#else
  asm volatile("s_wait_asynccnt 0" ::: "memory");
#endif
}
#endif

// ---------------------------------------------------------------------------
// GEMM: Y[b][m][n] = sum_k W[m][k] * X[b][k][n]
//   K = 192, N = 16384. Tile: 64(M) x 128(N) per workgroup, BK = 32.
//   8 waves arranged 4(M) x 2(N); each wave owns 16x64 -> 4 WMMA accumulators.
//   Fragment reads are wide LDS vector loads:
//     A elements [0..7]  = K hi*8..hi*8+7   (contiguous in Ws row)
//     A elements [8..15] = K 16+hi*8..+7    (contiguous in Ws row)
//     B elements [0..7] = K hi*8..+7, [8..15] = K 16+hi*8..+7 (XsT row)
// ---------------------------------------------------------------------------
template <typename TIn, typename TOut>
__global__ __launch_bounds__(256) void gemm_wx_kernel(
    const float* __restrict__ W, const TIn* __restrict__ Xall,
    TOut* __restrict__ Yall, int M, int K, int N) {
  const int b   = blockIdx.z;
  const TIn* X  = Xall + (size_t)b * K * N;
  TOut*      Y  = Yall + (size_t)b * M * N;
  const int n0  = blockIdx.x * 128;
  const int m0  = blockIdx.y * 64;
  const int tid = threadIdx.x;
  const int lane = tid & 31;
  const int wave = tid >> 5;
  const int wm = wave >> 1;  // 0..3
  const int wn = wave & 1;   // 0..1
  const int hi = (lane >> 4) & 1;
  const int l15 = lane & 15;

  // row stride 40 elems (80 B): multiple of 16 B (aligned b128 reads),
  // and 20-dword stride -> conflict-free bank rotation across 16 rows.
  __shared__ __bf16 Ws[64][40];    // [m][k], K contiguous
  __shared__ __bf16 XsT[128][40];  // [n][k], K contiguous (transposed tile)

  v8f acc[4];
#pragma unroll
  for (int i = 0; i < 4; ++i) acc[i] = zero8();

  // staging decomposition
  const int wr = tid >> 2;         // 0..63   W row
  const int wc = (tid & 3) * 8;    // 0,8,16,24  W col group
  const int xn = tid & 127;        // 0..127  X column -> XsT row
  const int xk = (tid >> 7) * 16;  // 0 | 16  K half

  for (int k0 = 0; k0 < K; k0 += 32) {
    // W tile 64x32: 8 consecutive K per thread (vectorizable both sides)
#pragma unroll
    for (int i = 0; i < 8; ++i)
      Ws[wr][wc + i] = (__bf16)W[(m0 + wr) * K + k0 + wc + i];
    // X tile 32x128 transposed: thread owns column xn, K-range [xk, xk+16);
    // global reads coalesced across threads (single-use stream -> NT hint),
    // LDS writes 32 B contiguous.
#pragma unroll
    for (int i = 0; i < 16; ++i)
      XsT[xn][xk + i] =
          (__bf16)__builtin_nontemporal_load(&X[(size_t)(k0 + xk + i) * N + (n0 + xn)]);
    __syncthreads();

    if (k0 + 32 < K)  // prefetch next X tile (global_prefetch_b8)
      __builtin_prefetch(&X[(size_t)(k0 + 32 + xk) * N + n0 + xn], 0, 1);

    // A fragment: two aligned 16B LDS reads
    const int am = wm * 16 + l15;
    v8bf a0 = *(const v8bf*)&Ws[am][hi * 8];
    v8bf a1 = *(const v8bf*)&Ws[am][16 + hi * 8];
    v16bf afrag = cat8(a0, a1);

#pragma unroll
    for (int nf = 0; nf < 4; ++nf) {
      const int bn = wn * 64 + nf * 16 + l15;
      v8bf b0 = *(const v8bf*)&XsT[bn][hi * 8];
      v8bf b1 = *(const v8bf*)&XsT[bn][16 + hi * 8];
      v16bf bfrag = cat8(b0, b1);
      acc[nf] = __builtin_amdgcn_wmma_f32_16x16x32_bf16(
          false, afrag, false, bfrag, (short)0, acc[nf], false, false);
    }
    __syncthreads();
  }

  // store: C/D layout M = r + 8*hi, N = lane&15 (coalesced across lanes)
#pragma unroll
  for (int nf = 0; nf < 4; ++nf)
#pragma unroll
    for (int r = 0; r < 8; ++r) {
      int row = m0 + wm * 16 + r + (hi << 3);
      int col = n0 + wn * 64 + nf * 16 + l15;
      Y[(size_t)row * N + col] = (TOut)acc[nf][r];
    }
}

// ---------------------------------------------------------------------------
// Depthwise 3x3 conv (padding=1) + fused sum-of-squares for q/k L2 norms.
// Each block: one (b,ch) plane, 2 output rows. The 4 needed input rows
// (1 KB, contiguous) are async-staged into LDS; compute reads LDS with
// 3-4x reuse. Block reduction -> global_atomic_add_f32 for norms.
// ---------------------------------------------------------------------------
__global__ __launch_bounds__(256) void dwconv_kernel(
    const __bf16* __restrict__ qkv, const float* __restrict__ wdw,
    __bf16* __restrict__ qkvd, float* __restrict__ sumsq) {
  const int ch = blockIdx.y;   // 0..575
  const int b  = blockIdx.z;   // 0..7
  const size_t base = ((size_t)b * 576 + ch) * 16384;
  const int y0 = blockIdx.x * 2;  // two output rows per block
  const int tid = threadIdx.x;

  __shared__ __align__(16) __bf16 tile[4][128];  // rows y0-1 .. y0+2 (clamped)

#if HAVE_ASYNC_LDS
  if (tid < 64) {
    int r = tid >> 4;          // 0..3
    int c = (tid & 15) * 8;    // 16B chunk start
    int sr = y0 - 1 + r;
    sr = sr < 0 ? 0 : (sr > 127 ? 127 : sr);  // clamped halo (unused rows)
    async_copy16(qkv + base + sr * 128 + c, &tile[r][c]);
  }
  async_wait0();
#else
  for (int i = tid; i < 512; i += 256) {
    int r = i >> 7, c = i & 127;
    int sr = y0 - 1 + r;
    sr = sr < 0 ? 0 : (sr > 127 ? 127 : sr);
    tile[r][c] = qkv[base + sr * 128 + c];
  }
#endif
  __syncthreads();

  const int ly = tid >> 7;   // 0 or 1
  const int x  = tid & 127;
  const int y  = y0 + ly;

  float wloc[9];
#pragma unroll
  for (int i = 0; i < 9; ++i) wloc[i] = wdw[ch * 9 + i];

  float acc = 0.f;
#pragma unroll
  for (int dy = -1; dy <= 1; ++dy) {
    int yy = y + dy;
    if (yy < 0 || yy > 127) continue;
#pragma unroll
    for (int dx = -1; dx <= 1; ++dx) {
      int xx = x + dx;
      if (xx < 0 || xx > 127) continue;
      acc += wloc[(dy + 1) * 3 + (dx + 1)] * (float)tile[ly + 1 + dy][xx];
    }
  }
  qkvd[base + y * 128 + x] = (__bf16)acc;

  __shared__ float red[256];
  red[tid] = (ch < 384) ? acc * acc : 0.f;
  __syncthreads();
  for (int s = 128; s > 0; s >>= 1) {
    if (tid < s) red[tid] += red[tid + s];
    __syncthreads();
  }
  if (tid == 0 && ch < 384) {
    int which = (ch < 192) ? 0 : 1;
    int c = ch - which * 192;
    atomicAdd(&sumsq[(which * 8 + b) * 192 + c], red[0]);
  }
}

__global__ void zero_kernel(float* p, int n) {
  int i = blockIdx.x * 256 + threadIdx.x;
  if (i < n) p[i] = 0.f;
}

// ---------------------------------------------------------------------------
// score: per (b, head) compute S = q . k^T over spatial (K=16384), then
// scale by 1/(|q_i||k_j|) * temperature, softmax rows, store bf16 48x64
// (zero-padded K for the AV GEMM). 8 waves split K; partials merged with
// ds_add_f32 atomics into LDS.
// ---------------------------------------------------------------------------
__global__ __launch_bounds__(256) void score_kernel(
    const __bf16* __restrict__ qkvd, const float* __restrict__ sumsq,
    const float* __restrict__ temperature, __bf16* __restrict__ score) {
  const int head = blockIdx.x;  // 0..3
  const int b    = blockIdx.y;  // 0..7
  const int tid  = threadIdx.x;
  const int lane = tid & 31;
  const int wave = tid >> 5;    // 0..7
  const int hi = (lane >> 4) & 1;
  const int l15 = lane & 15;

  const size_t qbase = ((size_t)b * 576 + head * 48) * 16384;
  const size_t kbase = ((size_t)b * 576 + 192 + head * 48) * 16384;

  __shared__ float S[48][48];
  float* Sf = &S[0][0];
  for (int i = tid; i < 48 * 48; i += 256) Sf[i] = 0.f;
  __syncthreads();

  v8f acc[3][3];
#pragma unroll
  for (int i = 0; i < 3; ++i)
#pragma unroll
    for (int j = 0; j < 3; ++j) acc[i][j] = zero8();

  const int s0 = wave * 2048;
  for (int ks = 0; ks < 2048; ks += 32) {
    const int kb = s0 + ks;
    v16bf a[3], bb[3];
#pragma unroll
    for (int t = 0; t < 3; ++t) {
      const int row = t * 16 + l15;  // A: M row; B: N column (same index math)
#pragma unroll
      for (int v = 0; v < 8; ++v)
#pragma unroll
        for (int h = 0; h < 2; ++h) {
          int kkA = ((v & 4) ? 16 : 0) + (hi << 3) + ((v & 3) << 1) + h;
          a[t][2 * v + h]  = qkvd[qbase + (size_t)row * 16384 + kb + kkA];
          int kkB = ((v & 4) ? 16 : 0) + (hi << 3) + ((v & 3) << 1) + h;
          bb[t][2 * v + h] = qkvd[kbase + (size_t)row * 16384 + kb + kkB];
        }
    }
#pragma unroll
    for (int mt = 0; mt < 3; ++mt)
#pragma unroll
      for (int nt = 0; nt < 3; ++nt)
        acc[mt][nt] = __builtin_amdgcn_wmma_f32_16x16x32_bf16(
            false, a[mt], false, bb[nt], (short)0, acc[mt][nt], false, false);
  }

  // merge partial 48x48 sums into LDS (ds_add_f32)
#pragma unroll
  for (int mt = 0; mt < 3; ++mt)
#pragma unroll
    for (int nt = 0; nt < 3; ++nt)
#pragma unroll
      for (int r = 0; r < 8; ++r) {
        int row = mt * 16 + r + (hi << 3);
        int col = nt * 16 + l15;
        atomicAdd(&S[row][col], acc[mt][nt][r]);
      }
  __syncthreads();

  // normalized, temperature-scaled softmax per row
  if (tid < 48) {
    const float* sq_q = sumsq + (0 * 8 + b) * 192 + head * 48;
    const float* sq_k = sumsq + (1 * 8 + b) * 192 + head * 48;
    float rq = 1.f / fmaxf(sqrtf(sq_q[tid]), 1e-12f);
    float tmp = temperature[head];
    float vals[48];
    float mx = -3.4e38f;
#pragma unroll
    for (int j = 0; j < 48; ++j) {
      float rk = 1.f / fmaxf(sqrtf(sq_k[j]), 1e-12f);
      float v = S[tid][j] * rq * rk * tmp;
      vals[j] = v;
      mx = fmaxf(mx, v);
    }
    float sum = 0.f;
#pragma unroll
    for (int j = 0; j < 48; ++j) {
      vals[j] = __expf(vals[j] - mx);
      sum += vals[j];
    }
    float inv = 1.f / sum;
    __bf16* out = score + ((size_t)(b * 4 + head)) * 48 * 64 + (size_t)tid * 64;
#pragma unroll
    for (int j = 0; j < 48; ++j) out[j] = (__bf16)(vals[j] * inv);
#pragma unroll
    for (int j = 48; j < 64; ++j) out[j] = (__bf16)0.f;
  }
}

// ---------------------------------------------------------------------------
// AV: out = S(48x64, K-padded) x v(48, 16384) per (b, head).
// Score matrix (6 KB) is async-staged into LDS once per workgroup (all 8
// waves reuse it); A-fragments are aligned 16B LDS vector reads. Each wave
// owns a 16-column slice; 3 M-tiles; 2 K-steps.
// ---------------------------------------------------------------------------
__global__ __launch_bounds__(256) void av_kernel(
    const __bf16* __restrict__ score, const __bf16* __restrict__ qkvd,
    __bf16* __restrict__ attn) {
  const int head = blockIdx.y;  // 0..3
  const int b    = blockIdx.z;  // 0..7
  const int tid  = threadIdx.x;
  const int lane = tid & 31;
  const int wave = tid >> 5;    // 0..7
  const int hi = (lane >> 4) & 1;
  const int l15 = lane & 15;
  const int n0 = blockIdx.x * 128 + wave * 16;

  const __bf16* sc = score + ((size_t)(b * 4 + head)) * 48 * 64;
  const size_t vbase = ((size_t)b * 576 + 384 + head * 48) * 16384;

  // 48 rows x 64 K, padded to 72-elem rows (144 B: 16B-aligned, bank-rotated)
  __shared__ __align__(16) __bf16 scs[48 * 72];
#if HAVE_ASYNC_LDS
  for (int c = tid; c < 384; c += 256) {  // 384 chunks of 16 B
    int row = c >> 3, wi = (c & 7) * 8;
    async_copy16((const char*)sc + (row * 64 + wi) * 2, &scs[row * 72 + wi]);
  }
  async_wait0();
#else
  for (int c = tid; c < 384; c += 256) {
    int row = c >> 3, wi = (c & 7) * 8;
    *(v8bf*)&scs[row * 72 + wi] = *(const v8bf*)(sc + row * 64 + wi);
  }
#endif
  __syncthreads();

  v8f acc[3];
#pragma unroll
  for (int i = 0; i < 3; ++i) acc[i] = zero8();

#pragma unroll
  for (int kstep = 0; kstep < 2; ++kstep) {
    // B fragment: K = channel j of v, N = spatial column
    v16bf bfrag;
#pragma unroll
    for (int v = 0; v < 8; ++v)
#pragma unroll
      for (int h = 0; h < 2; ++h) {
        int kk = kstep * 32 + ((v & 4) ? 16 : 0) + (hi << 3) + ((v & 3) << 1) + h;
        int j = kk > 47 ? 47 : kk;  // pad region: A is zero there anyway
        bfrag[2 * v + h] = qkvd[vbase + (size_t)j * 16384 + n0 + l15];
      }
#pragma unroll
    for (int mt = 0; mt < 3; ++mt) {
      const int row = mt * 16 + l15;
      v8bf a0 = *(const v8bf*)&scs[row * 72 + kstep * 32 + hi * 8];
      v8bf a1 = *(const v8bf*)&scs[row * 72 + kstep * 32 + 16 + hi * 8];
      v16bf afrag = cat8(a0, a1);
      acc[mt] = __builtin_amdgcn_wmma_f32_16x16x32_bf16(
          false, afrag, false, bfrag, (short)0, acc[mt], false, false);
    }
  }
#pragma unroll
  for (int mt = 0; mt < 3; ++mt)
#pragma unroll
    for (int r = 0; r < 8; ++r) {
      int row = mt * 16 + r + (hi << 3);
      int col = n0 + l15;
      attn[((size_t)b * 192 + head * 48 + row) * 16384 + col] = (__bf16)acc[mt][r];
    }
}

// ---------------------------------------------------------------------------
extern "C" void kernel_launch(void* const* d_in, const int* in_sizes, int n_in,
                              void* d_out, int out_size, void* d_ws,
                              size_t ws_size, hipStream_t stream) {
  (void)in_sizes; (void)n_in; (void)out_size; (void)ws_size;
  const float* x           = (const float*)d_in[0];  // [8,192,128,128]
  const float* w_qkv       = (const float*)d_in[1];  // [576,192]
  const float* w_dw        = (const float*)d_in[2];  // [576,1,3,3]
  const float* w_proj      = (const float*)d_in[3];  // [192,192]
  const float* temperature = (const float*)d_in[4];  // [4]
  float* out = (float*)d_out;                        // [8,192,128,128]

  // workspace layout (bytes)
  char* ws = (char*)d_ws;
  __bf16* qkv   = (__bf16*)(ws);                 // 8*576*16384*2 = 150,994,944
  __bf16* qkvd  = (__bf16*)(ws + 150994944);     // 150,994,944
  __bf16* attn  = (__bf16*)(ws + 301989888);     // 8*192*16384*2 = 50,331,648
  __bf16* score = (__bf16*)(ws + 352321536);     // 8*4*48*64*2   = 196,608
  float*  sumsq = (float*)(ws + 352518144);      // 2*8*192*4     = 12,288

  zero_kernel<<<dim3(12), 256, 0, stream>>>(sumsq, 3072);

  // qkv = w_qkv @ x  (per batch), bf16 out
  gemm_wx_kernel<float, __bf16>
      <<<dim3(128, 9, 8), 256, 0, stream>>>(w_qkv, x, qkv, 576, 192, 16384);

  // depthwise 3x3 + q/k sum-of-squares
  dwconv_kernel<<<dim3(64, 576, 8), 256, 0, stream>>>(qkv, w_dw, qkvd, sumsq);

  // softmax( (q/|q|) (k/|k|)^T * temp )  -> bf16 48x64 per (b,head)
  score_kernel<<<dim3(4, 8), 256, 0, stream>>>(qkvd, sumsq, temperature, score);

  // out_head = score @ v
  av_kernel<<<dim3(128, 4, 8), 256, 0, stream>>>(score, qkvd, attn);

  // final projection: out = w_proj @ attn (per batch), f32 out
  gemm_wx_kernel<__bf16, float>
      <<<dim3(128, 3, 8), 256, 0, stream>>>(w_proj, attn, out, 192, 192, 16384);
}
